// HNNModel_81535659147708
// MI455X (gfx1250) — matrix operand order repaired and checked by reading, not compiled, not run on hardware
//
#include <hip/hip_runtime.h>
#include <hip/hip_bf16.h>

typedef __bf16 v16bf __attribute__((ext_vector_type(16)));
typedef float  v8f   __attribute__((ext_vector_type(8)));

#if __has_builtin(__builtin_amdgcn_tensor_load_to_lds)
#define HAVE_TDM 1
typedef unsigned int v4u __attribute__((ext_vector_type(4)));
typedef int v8i __attribute__((ext_vector_type(8)));
typedef int v4i __attribute__((ext_vector_type(4)));
#else
#define HAVE_TDM 0
#endif

#define DECAY_F  0.6f
#define THRESH_F 0.6f

constexpr int T = 64, B = 32;
constexpr int NTOKEN = 32000;
constexpr int NINP = 1024;
constexpr int H = 2048;          // S2 + R2 (= o1/o2 width)
constexpr int CH = 512;          // elements per k-chunk: 32 lanes * 16 bf16

// ---------------------------------------------------------------------------
// WMMA helper: D = A(16x32 bf16) * B(32x16 bf16) + C(16x16 f32)
// ---------------------------------------------------------------------------
__device__ __forceinline__ v8f wmma_bf16(v16bf a, v16bf b, v8f c) {
  return __builtin_amdgcn_wmma_f32_16x16x32_bf16(false, a, false, b,
                                                 (short)0, c, false, false);
}

__device__ __forceinline__ float sigmoidf(float x) {
  return 1.0f / (1.0f + __expf(-x));
}

// A-matrix (16x32 bf16) per-lane k offsets: lanes 0-15 hold k=0..15,
// lanes 16-31 hold k=16..31 (matches B layout).
__device__ __forceinline__ int a_koff(int lane, int j) {
  return ((lane >= 16) ? 16 : 0) + j;
}
// B-matrix (32x16 bf16): lanes 0-15 hold k=0..15, lanes 16-31 hold k=16..31.
__device__ __forceinline__ int b_koff(int lane, int j) {
  return ((lane >= 16) ? 16 : 0) + j;
}

__device__ __forceinline__ void split_bf16(float v, __bf16& hi, __bf16& lo) {
  hi = (__bf16)v;
  lo = (__bf16)(v - (float)hi);
}

// ---------------------------------------------------------------------------
// Embedding gather + hi/lo split into WMMA A layout.
// dst layout: [(t*2+mt)][kc 0..31][lane][16]
// ---------------------------------------------------------------------------
__global__ void __launch_bounds__(256)
embed_pack_kernel(const int* __restrict__ raw, const float* __restrict__ enc,
                  __bf16* __restrict__ xhi, __bf16* __restrict__ xlo) {
  int tid  = blockIdx.x * blockDim.x + threadIdx.x;  // T*2*32*32 = 131072
  int lane = tid & 31;
  int kc   = (tid >> 5) & 31;
  int mt   = (tid >> 10) & 1;
  int t    = tid >> 11;
  if (t >= T) return;
  int brow = mt * 16 + (lane & 15);
  int tok  = raw[t * B + brow];
  const float* src = enc + (size_t)tok * NINP;
  size_t dst = ((size_t)((t * 2 + mt) * 32 + kc) * 32 + lane) * 16;
#pragma unroll
  for (int j = 0; j < 16; ++j) {
    float v = src[kc * 32 + a_koff(lane, j)];
    split_bf16(v, xhi[dst + j], xlo[dst + j]);
  }
}

// ---------------------------------------------------------------------------
// Weight pack: W[K][1024] f32 row-major -> B layout [nt 0..63][kc][lane][16]
// ---------------------------------------------------------------------------
__global__ void __launch_bounds__(256)
packB_kernel(const float* __restrict__ W, int K,
             __bf16* __restrict__ dhi, __bf16* __restrict__ dlo) {
  int tid  = blockIdx.x * blockDim.x + threadIdx.x;  // 64*(K/32)*32
  int nkc  = K >> 5;
  int lane = tid & 31;
  int kc   = (tid >> 5) % nkc;
  int nt   = (tid >> 5) / nkc;
  if (nt >= 64) return;
  int n = nt * 16 + (lane & 15);
  size_t dst = ((size_t)(nt * nkc + kc) * 32 + lane) * 16;
#pragma unroll
  for (int j = 0; j < 16; ++j) {
    int k = kc * 32 + b_koff(lane, j);
    float v = W[(size_t)k * 1024 + n];
    split_bf16(v, dhi[dst + j], dlo[dst + j]);
  }
}

// ---------------------------------------------------------------------------
// Activation pack: src f32 [32 x srcW] -> A layout
// dst index = ((mt*mtStride + kcBase + kc)*32 + lane)*16 + j ; k from colBase
// dlo == nullptr -> hi only (decoder A).
// ---------------------------------------------------------------------------
__global__ void __launch_bounds__(256)
packA_kernel(const float* __restrict__ src, int srcW, int colBase, int nkc,
             __bf16* __restrict__ dhi, __bf16* __restrict__ dlo,
             int mtStride, int kcBase) {
  int tid  = blockIdx.x * blockDim.x + threadIdx.x;  // 2*nkc*32
  int lane = tid & 31;
  int kc   = (tid >> 5) % nkc;
  int mt   = (tid >> 5) / nkc;
  if (mt >= 2) return;
  int m = mt * 16 + (lane & 15);
  size_t dst = ((size_t)(mt * mtStride + kcBase + kc) * 32 + lane) * 16;
#pragma unroll
  for (int j = 0; j < 16; ++j) {
    float v = src[(size_t)m * srcW + colBase + kc * 32 + a_koff(lane, j)];
    __bf16 h = (__bf16)v;
    dhi[dst + j] = h;
    if (dlo) dlo[dst + j] = (__bf16)(v - (float)h);
  }
}

// ---------------------------------------------------------------------------
// One recurrent half-step (shared by layer1/layer2).
//   waves 0..127 : S-path  acc = x @ Ws           -> membrane/spike update
//   waves 128..255: R-path acc = x @ Wr + y @ Wv  -> sigmoid update
// All branching wave-uniform -> EXEC all ones for WMMA.
// ---------------------------------------------------------------------------
__global__ void __launch_bounds__(256)
phase_kernel(const __bf16* __restrict__ Axhi, const __bf16* __restrict__ Axlo,
             int x_nkc, int x_mtStride,
             const __bf16* __restrict__ Ayhi, const __bf16* __restrict__ Aylo,
             int y_mtStride, int y_kcBase,
             const __bf16* __restrict__ Wshi, const __bf16* __restrict__ Wslo,
             const __bf16* __restrict__ Wrhi, const __bf16* __restrict__ Wrlo,
             const __bf16* __restrict__ Wvhi, const __bf16* __restrict__ Wvlo,
             float* __restrict__ mem, float* __restrict__ spike,
             float* __restrict__ yout, float* __restrict__ oout) {
  int wid  = (blockIdx.x * blockDim.x + threadIdx.x) >> 5;  // 0..255
  int lane = threadIdx.x & 31;
  bool isS = wid < 128;
  int g  = isS ? wid : (wid - 128);
  int nt = g >> 1;
  int mt = g & 1;
  int n     = nt * 16 + (lane & 15);
  int mrow0 = mt * 16 + ((lane >= 16) ? 8 : 0);
  v8f acc = {};

  if (isS) {
    const __bf16* wh = Wshi + (size_t)nt * x_nkc * CH;
    const __bf16* wl = Wslo + (size_t)nt * x_nkc * CH;
    for (int kc = 0; kc < x_nkc; ++kc) {
      size_t ai = ((size_t)(mt * x_mtStride + kc) * 32 + lane) * 16;
      size_t bi = ((size_t)kc * 32 + lane) * 16;
      v16bf ah = *(const v16bf*)(Axhi + ai);
      v16bf al = *(const v16bf*)(Axlo + ai);
      v16bf bh = *(const v16bf*)(wh + bi);
      v16bf bl = *(const v16bf*)(wl + bi);
      acc = wmma_bf16(ah, bh, acc);
      acc = wmma_bf16(ah, bl, acc);
      acc = wmma_bf16(al, bh, acc);
    }
#pragma unroll
    for (int i = 0; i < 8; ++i) {
      int m = mrow0 + i;
      size_t idx = (size_t)m * 1024 + n;
      float hm = mem[idx];
      float sp = spike[idx];
      hm = hm * (1.0f - sp) * DECAY_F + acc[i];
      float nsp = (hm > THRESH_F) ? 1.0f : 0.0f;
      mem[idx]   = hm;
      spike[idx] = nsp;
      oout[(size_t)m * H + n] = nsp;
    }
  } else {
    const __bf16* wh = Wrhi + (size_t)nt * x_nkc * CH;
    const __bf16* wl = Wrlo + (size_t)nt * x_nkc * CH;
    for (int kc = 0; kc < x_nkc; ++kc) {
      size_t ai = ((size_t)(mt * x_mtStride + kc) * 32 + lane) * 16;
      size_t bi = ((size_t)kc * 32 + lane) * 16;
      v16bf ah = *(const v16bf*)(Axhi + ai);
      v16bf al = *(const v16bf*)(Axlo + ai);
      v16bf bh = *(const v16bf*)(wh + bi);
      v16bf bl = *(const v16bf*)(wl + bi);
      acc = wmma_bf16(ah, bh, acc);
      acc = wmma_bf16(ah, bl, acc);
      acc = wmma_bf16(al, bh, acc);
    }
    const __bf16* vh = Wvhi + (size_t)nt * 32 * CH;
    const __bf16* vl = Wvlo + (size_t)nt * 32 * CH;
    for (int kc = 0; kc < 32; ++kc) {
      size_t ai = ((size_t)(mt * y_mtStride + y_kcBase + kc) * 32 + lane) * 16;
      size_t bi = ((size_t)kc * 32 + lane) * 16;
      v16bf ah = *(const v16bf*)(Ayhi + ai);
      v16bf al = *(const v16bf*)(Aylo + ai);
      v16bf bh = *(const v16bf*)(vh + bi);
      v16bf bl = *(const v16bf*)(vl + bi);
      acc = wmma_bf16(ah, bh, acc);
      acc = wmma_bf16(ah, bl, acc);
      acc = wmma_bf16(al, bh, acc);
    }
#pragma unroll
    for (int i = 0; i < 8; ++i) {
      int m = mrow0 + i;
      float y = sigmoidf(acc[i]);
      yout[(size_t)m * 1024 + n]     = y;
      oout[(size_t)m * H + 1024 + n] = y;
    }
  }
}

// ---------------------------------------------------------------------------
// Decoder: decoded[m][v] = outs[m][:] . decoder_w[v][:] + b[v]
// Block = one 16-col n-tile x 1024-row m-chunk; 8 waves x 8 m-tiles.
// B slice (16 rows x 2048 f32 = 128 KB of decoder_w) DMA'd into dynamic LDS
// once per block via the Tensor Data Mover (tensor_load_to_lds +
// s_wait_tensorcnt); no per-kc barriers. Lanes convert f32->bf16 in-register.
// ---------------------------------------------------------------------------
__global__ void __launch_bounds__(256)
decoder_kernel(const __bf16* __restrict__ Adec, const float* __restrict__ Wd,
               const float* __restrict__ bias, float* __restrict__ out) {
  extern __shared__ float Bf[];     // [16][2048] f32, 128 KB dynamic LDS
  int nt     = blockIdx.x >> 1;     // 0..1999
  int mchunk = blockIdx.x & 1;      // 0..1
  int wid    = threadIdx.x >> 5;    // 0..7
  int lane   = threadIdx.x & 31;
  int n0     = nt * 16;

#if HAVE_TDM
  if (wid == 0) {
    // TDM descriptor: 2D tile, 16 rows x 2048 f32, row stride 2048 f32.
    unsigned long long ga =
        (unsigned long long)(const void*)(Wd + (size_t)n0 * 2048);
    v4u g0;
    g0[0] = 1u;                                   // count = 1 (valid D#)
    g0[1] = 0u;                                   // lds_addr = dynamic-LDS base
    g0[2] = (unsigned)(ga & 0xffffffffu);         // global_addr[31:0]
    g0[3] = (unsigned)((ga >> 32) & 0x01ffffffu)  // global_addr[56:32]
            | (2u << 30);                         // type = 2 (image)
    v8i g1;
    g1[0] = (int)(2u << 16);                      // data_size = 4 bytes
    g1[1] = (int)(2048u << 16);                   // tensor_dim0 = 2048
    g1[2] = (int)(32000u << 16);                  // tensor_dim1 = 32000
    g1[3] = (int)(2048u << 16);                   // tile_dim0 = 2048
    g1[4] = 16;                                   // tile_dim1 = 16
    g1[5] = 2048;                                 // tensor_dim0_stride = 2048
    g1[6] = 0;
    g1[7] = 0;
    v4i g2 = {0, 0, 0, 0};
    v4i g3 = {0, 0, 0, 0};
    v8i g4 = {0, 0, 0, 0, 0, 0, 0, 0};            // extra group (6-arg form)
    __builtin_amdgcn_tensor_load_to_lds(g0, g1, g2, g3, g4, 0);
    __builtin_amdgcn_s_wait_tensorcnt((short)0);
  }
  __syncthreads();
#else
  // Fallback: cooperative fill of the same LDS layout.
  for (int i = threadIdx.x; i < 16 * 512; i += 256) {
    ((float4*)Bf)[i] =
        *(const float4*)(Wd + (size_t)(n0 + (i >> 9)) * 2048 + ((i & 511) << 2));
  }
  __syncthreads();
#endif

  v8f acc[8];
#pragma unroll
  for (int i = 0; i < 8; ++i) acc[i] = {};

  int r    = lane & 15;              // decoder_w row within tile (= local n)
  int half = (lane >= 16) ? 16 : 0;  // k-half per B layout

  for (int kc = 0; kc < 64; ++kc) {
    const float4* lp = (const float4*)&Bf[(size_t)r * 2048 + kc * 32 + half];
    float4 f0 = lp[0], f1 = lp[1], f2 = lp[2], f3 = lp[3];
    float fv[16] = {f0.x, f0.y, f0.z, f0.w, f1.x, f1.y, f1.z, f1.w,
                    f2.x, f2.y, f2.z, f2.w, f3.x, f3.y, f3.z, f3.w};
    v16bf b;
#pragma unroll
    for (int j = 0; j < 16; ++j) b[j] = (__bf16)fv[j];

    // prefetch A stream two chunks ahead
    __builtin_prefetch(
        Adec + ((size_t)((mchunk * 64 + wid * 8) * 64 + kc + 2) * 32 + lane) * 16,
        0, 1);
#pragma unroll
    for (int mi = 0; mi < 8; ++mi) {
      int mtg = mchunk * 64 + wid * 8 + mi;
      v16bf a = *(const v16bf*)(Adec + ((size_t)(mtg * 64 + kc) * 32 + lane) * 16);
      acc[mi] = wmma_bf16(a, b, acc[mi]);
    }
  }

  float bv = bias[n0 + (lane & 15)];
  int n = n0 + (lane & 15);
#pragma unroll
  for (int mi = 0; mi < 8; ++mi) {
    int mtg = mchunk * 64 + wid * 8 + mi;
    int m0  = mtg * 16 + ((lane >= 16) ? 8 : 0);
#pragma unroll
    for (int i = 0; i < 8; ++i) {
      out[(size_t)(m0 + i) * NTOKEN + n] = acc[mi][i] + bv;
    }
  }
}

// ---------------------------------------------------------------------------
__global__ void __launch_bounds__(256)
zero4_kernel(float* a, float* b, float* c, float* d) {
  int i = blockIdx.x * blockDim.x + threadIdx.x;
  if (i < 32768) { a[i] = 0.0f; b[i] = 0.0f; c[i] = 0.0f; d[i] = 0.0f; }
}

__global__ void __launch_bounds__(256)
final_copy_kernel(const float* __restrict__ h1y, const float* __restrict__ h2y,
                  float* __restrict__ out) {
  int i = blockIdx.x * blockDim.x + threadIdx.x;
  if (i < 32768) {
    out[(size_t)T * B * NTOKEN + i]         = h1y[i];
    out[(size_t)T * B * NTOKEN + 32768 + i] = h2y[i];
  }
}

// ---------------------------------------------------------------------------
extern "C" void kernel_launch(void* const* d_in, const int* in_sizes, int n_in,
                              void* d_out, int out_size, void* d_ws, size_t ws_size,
                              hipStream_t stream) {
  const int*   raw   = (const int*)  d_in[0];   // [64,32]
  const float* h1y0  = (const float*)d_in[1];   // [32,1024]
  const float* h2y0  = (const float*)d_in[2];   // [32,1024]
  const float* enc   = (const float*)d_in[3];   // [32000,1024]
  const float* w_s1  = (const float*)d_in[4];   // snn_fc1 [1024,1024]
  const float* w_s2  = (const float*)d_in[5];   // snn_fc2 [2048,1024]
  const float* w_r1  = (const float*)d_in[6];   // rnn_fc1 [1024,1024]
  const float* w_r2  = (const float*)d_in[7];   // rnn_fc2 [2048,1024]
  const float* w_v1  = (const float*)d_in[8];   // rnn_fv1 [1024,1024]
  const float* w_v2  = (const float*)d_in[9];   // rnn_fv2 [1024,1024]
  const float* w_dec = (const float*)d_in[10];  // [32000,2048]
  const float* b_dec = (const float*)d_in[11];  // [32000]
  float* out = (float*)d_out;

  // workspace bump allocator (256B aligned)
  char* ws = (char*)d_ws;
  auto alloc = [&](size_t bytes) -> char* {
    char* p = ws;
    ws += (bytes + 255) & ~(size_t)255;
    return p;
  };
  __bf16* xhi   = (__bf16*)alloc((size_t)T * 2 * 32 * CH * 2);
  __bf16* xlo   = (__bf16*)alloc((size_t)T * 2 * 32 * CH * 2);
  __bf16* w1sh  = (__bf16*)alloc((size_t)64 * 32 * CH * 2);
  __bf16* w1sl  = (__bf16*)alloc((size_t)64 * 32 * CH * 2);
  __bf16* w1rh  = (__bf16*)alloc((size_t)64 * 32 * CH * 2);
  __bf16* w1rl  = (__bf16*)alloc((size_t)64 * 32 * CH * 2);
  __bf16* w1vh  = (__bf16*)alloc((size_t)64 * 32 * CH * 2);
  __bf16* w1vl  = (__bf16*)alloc((size_t)64 * 32 * CH * 2);
  __bf16* w2sh  = (__bf16*)alloc((size_t)64 * 64 * CH * 2);
  __bf16* w2sl  = (__bf16*)alloc((size_t)64 * 64 * CH * 2);
  __bf16* w2rh  = (__bf16*)alloc((size_t)64 * 64 * CH * 2);
  __bf16* w2rl  = (__bf16*)alloc((size_t)64 * 64 * CH * 2);
  __bf16* w2vh  = (__bf16*)alloc((size_t)64 * 32 * CH * 2);
  __bf16* w2vl  = (__bf16*)alloc((size_t)64 * 32 * CH * 2);
  __bf16* o1ph  = (__bf16*)alloc((size_t)2 * 64 * CH * 2);
  __bf16* o1pl  = (__bf16*)alloc((size_t)2 * 64 * CH * 2);
  __bf16* h2yph = (__bf16*)alloc((size_t)2 * 32 * CH * 2);
  __bf16* h2ypl = (__bf16*)alloc((size_t)2 * 32 * CH * 2);
  __bf16* decA  = (__bf16*)alloc((size_t)128 * 64 * CH * 2);
  float*  o1f   = (float*) alloc((size_t)32 * H * 4);
  float*  o2f   = (float*) alloc((size_t)32 * H * 4);
  float*  h1mem = (float*) alloc((size_t)32 * 1024 * 4);
  float*  h1sp  = (float*) alloc((size_t)32 * 1024 * 4);
  float*  h1y   = (float*) alloc((size_t)32 * 1024 * 4);
  float*  h2mem = (float*) alloc((size_t)32 * 1024 * 4);
  float*  h2sp  = (float*) alloc((size_t)32 * 1024 * 4);
  float*  h2y   = (float*) alloc((size_t)32 * 1024 * 4);

  // ---- init ----
  zero4_kernel<<<128, 256, 0, stream>>>(h1mem, h1sp, h2mem, h2sp);
  packA_kernel<<<8, 256, 0, stream>>>(h1y0, 1024, 0, 32, o1ph, o1pl, 64, 32);
  packA_kernel<<<8, 256, 0, stream>>>(h2y0, 1024, 0, 32, h2yph, h2ypl, 32, 0);
  packB_kernel<<<256, 256, 0, stream>>>(w_s1, 1024, w1sh, w1sl);
  packB_kernel<<<256, 256, 0, stream>>>(w_r1, 1024, w1rh, w1rl);
  packB_kernel<<<256, 256, 0, stream>>>(w_v1, 1024, w1vh, w1vl);
  packB_kernel<<<512, 256, 0, stream>>>(w_s2, 2048, w2sh, w2sl);
  packB_kernel<<<512, 256, 0, stream>>>(w_r2, 2048, w2rh, w2rl);
  packB_kernel<<<256, 256, 0, stream>>>(w_v2, 1024, w2vh, w2vl);
  embed_pack_kernel<<<512, 256, 0, stream>>>(raw, enc, xhi, xlo);

  // ---- recurrent loop ----
  for (int t = 0; t < T; ++t) {
    size_t xoff = (size_t)t * 2 * 32 * CH;
    phase_kernel<<<32, 256, 0, stream>>>(
        xhi + xoff, xlo + xoff, 32, 32,
        o1ph, o1pl, 64, 32,
        w1sh, w1sl, w1rh, w1rl, w1vh, w1vl,
        h1mem, h1sp, h1y, o1f);
    packA_kernel<<<16, 256, 0, stream>>>(o1f, H, 0, 64, o1ph, o1pl, 64, 0);
    phase_kernel<<<32, 256, 0, stream>>>(
        o1ph, o1pl, 64, 64,
        h2yph, h2ypl, 32, 0,
        w2sh, w2sl, w2rh, w2rl, w2vh, w2vl,
        h2mem, h2sp, h2y, o2f);
    packA_kernel<<<16, 256, 0, stream>>>(o2f, H, 0, 64,
                                         decA + (size_t)t * 2 * 64 * CH,
                                         nullptr, 64, 0);
    packA_kernel<<<8, 256, 0, stream>>>(o2f, H, 1024, 32, h2yph, h2ypl, 32, 0);
  }

  // ---- decoder GEMM: [2048 x 2048] x [2048 x 32000], 128 KB dynamic LDS ----
  decoder_kernel<<<4000, 256, 128 * 1024, stream>>>(decA, w_dec, b_dec, out);

  // ---- final states ----
  final_copy_kernel<<<128, 256, 0, stream>>>(h1y, h2y, out);
}
